// Attention_8358006358637
// MI455X (gfx1250) — matrix operand order repaired and checked by reading, not compiled, not run on hardware
//
#include <hip/hip_runtime.h>
#include <hip/hip_bf16.h>

typedef __attribute__((ext_vector_type(16))) _Float16 v16h;
typedef __attribute__((ext_vector_type(8)))  float    v8f;

#define B_        4
#define S_        1024
#define H_        16
#define KVH_      4
#define D_        64
#define SCALE_    0.125f
#define LOG2E_    1.4426950408889634f
#define NSLOTS_   8192
#define KVSTRIDE_ (KVH_ * D_)   // 256 floats per k/v row

// Native exp2: single v_exp_f32, no OCML denorm fixup. All our arguments are
// <= 0; hardware flush-to-zero below 2^-126 is the mathematically desired
// result (vanishing softmax probability).
__device__ __forceinline__ float fast_exp2(float x) {
  return __builtin_amdgcn_exp2f(x);
}

// ---------------------------------------------------------------------------
// KV-cache scatter: one block per token row, 256 threads copy the 256-float
// k row and v row into cache[slot].
// ---------------------------------------------------------------------------
__global__ __launch_bounds__(KVSTRIDE_) void scatter_kv_kernel(
    const float* __restrict__ k, const float* __restrict__ v,
    const int* __restrict__ slot_mapping,
    float* __restrict__ k_cache, float* __restrict__ v_cache) {
  const int row = blockIdx.x;
  const int t   = threadIdx.x;
  const int s   = slot_mapping[row];
  const size_t src = (size_t)row * KVSTRIDE_ + t;
  const size_t dst = (size_t)s   * KVSTRIDE_ + t;
  k_cache[dst] = k[src];
  v_cache[dst] = v[src];
}

// ---------------------------------------------------------------------------
// Causal GQA flash-attention, wave32 + v_wmma_f32_16x16x32_f16.
// One wave owns a 16-row query tile for one (batch, head). 8 waves / block.
// grid.x = B*H*(S/128). Softmax runs in base-2 (SCALE*log2e folded into Q).
// ---------------------------------------------------------------------------
__global__ __launch_bounds__(256) void attn_fwd_kernel(
    const float* __restrict__ q, const float* __restrict__ kmat,
    const float* __restrict__ vmat, float* __restrict__ out) {
  const int wave = threadIdx.x >> 5;
  const int lane = threadIdx.x & 31;
  const int lh   = lane & 15;   // row/col index within a 16-lane group
  const int lg   = lane >> 4;   // 0 = lanes 0-15, 1 = lanes 16-31

  const int chunks = S_ / 128;              // q-chunks per (b,h)
  const int bh = blockIdx.x / chunks;
  const int qc = blockIdx.x % chunks;
  const int b  = bh / H_;
  const int h  = bh % H_;
  const int kvh   = h / (H_ / KVH_);        // GQA mapping
  const int qbase = qc * 128 + wave * 16;   // first query row of this wave

  // Per-wave P staging buffer: 16 rows x 32 cols f16 (C-layout -> A-layout).
  __shared__ _Float16 pbuf[8 * 16 * 32];
  _Float16* pb = &pbuf[wave * (16 * 32)];

  // ---- Load Q tile as two f16 A-operands (SCALE*log2e folded in) ----
  // A order used here (contiguous k, same as B): lane (lh,lg): halves 0..7 -> K = lg*16+j,
  //                 halves 8..15 -> K = lg*16 + 8 + j.
  const float qfold = SCALE_ * LOG2E_;
  const float* qrow = q + (size_t)(b * S_ + qbase + lh) * (H_ * D_) + h * D_;
  v16h a0, a1;
#pragma unroll
  for (int j = 0; j < 8; ++j) {
    a0[j]     = (_Float16)(qrow[lg * 16 + j]      * qfold);
    a0[j + 8] = (_Float16)(qrow[8 + lg * 16 + j]  * qfold);
    a1[j]     = (_Float16)(qrow[32 + lg * 16 + j] * qfold);
    a1[j + 8] = (_Float16)(qrow[40 + lg * 16 + j] * qfold);
  }

  // Online-softmax state (log2 domain). C-layout: VGPR vv in this lane is
  // query row (qbase + vv + 8*lg), striped across the 16-lane group.
  v8f o0 = {}, o1 = {}, o2 = {}, o3 = {};
  float mrun[8], lrun[8];
#pragma unroll
  for (int vv = 0; vv < 8; ++vv) { mrun[vv] = -1e30f; lrun[vv] = 0.0f; }

  const float* kbase = kmat + (size_t)b * S_ * KVSTRIDE_ + kvh * D_;
  const float* vbase = vmat + (size_t)b * S_ * KVSTRIDE_ + kvh * D_;

  const int kend = qbase + 16;
  // Causal key loop, 32 keys per iteration; kb <= qbase always holds for the
  // final block, so every row sees >=1 unmasked key per block (no NaNs).
  for (int kb = 0; kb < kend; kb += 32) {
    // ---- K^T B-operands. B order used here (contiguous k, as in Q): lane (lh,lg) holds key column
    // lh, halves j -> reduction index lg*16 + j (16 contiguous dims). ----
    const float* kr0 = kbase + (size_t)(kb + lh)      * KVSTRIDE_ + lg * 16;
    const float* kr1 = kbase + (size_t)(kb + 16 + lh) * KVSTRIDE_ + lg * 16;
    v16h bk0lo, bk0hi, bk1lo, bk1hi;
#pragma unroll
    for (int j = 0; j < 16; ++j) {
      bk0lo[j] = (_Float16)kr0[j];
      bk0hi[j] = (_Float16)kr0[32 + j];
      bk1lo[j] = (_Float16)kr1[j];
      bk1hi[j] = (_Float16)kr1[32 + j];
    }

    v8f s0 = {}, s1 = {};
    s0 = __builtin_amdgcn_wmma_f32_16x16x32_f16(false, a0, false, bk0lo, (short)0, s0, false, false);
    s0 = __builtin_amdgcn_wmma_f32_16x16x32_f16(false, a1, false, bk0hi, (short)0, s0, false, false);
    s1 = __builtin_amdgcn_wmma_f32_16x16x32_f16(false, a0, false, bk1lo, (short)0, s1, false, false);
    s1 = __builtin_amdgcn_wmma_f32_16x16x32_f16(false, a1, false, bk1hi, (short)0, s1, false, false);

    // ---- Causal mask (only blocks that straddle the diagonal) ----
    if (kb + 31 > qbase) {
#pragma unroll
      for (int vv = 0; vv < 8; ++vv) {
        const int row = qbase + vv + 8 * lg;
        if (kb + lh      > row) s0[vv] = -3e38f;
        if (kb + 16 + lh > row) s1[vv] = -3e38f;
      }
    }

    // ---- Online softmax (base-2): row max / exp2 / row sum / rescale ----
#pragma unroll
    for (int vv = 0; vv < 8; ++vv) {
      float t = fmaxf(s0[vv], s1[vv]);
      t = fmaxf(t, __shfl_xor(t, 1, 32));
      t = fmaxf(t, __shfl_xor(t, 2, 32));
      t = fmaxf(t, __shfl_xor(t, 4, 32));
      t = fmaxf(t, __shfl_xor(t, 8, 32));
      const float mnew  = fmaxf(mrun[vv], t);
      const float scale = fast_exp2(mrun[vv] - mnew);
      const float p0 = fast_exp2(s0[vv] - mnew);
      const float p1 = fast_exp2(s1[vv] - mnew);
      float ps = p0 + p1;
      ps += __shfl_xor(ps, 1, 32);
      ps += __shfl_xor(ps, 2, 32);
      ps += __shfl_xor(ps, 4, 32);
      ps += __shfl_xor(ps, 8, 32);
      lrun[vv] = lrun[vv] * scale + ps;
      mrun[vv] = mnew;
      o0[vv] *= scale; o1[vv] *= scale; o2[vv] *= scale; o3[vv] *= scale;
      // C-layout -> LDS (row-major 16x32 f16)
      pb[(vv + 8 * lg) * 32 + lh]      = (_Float16)p0;
      pb[(vv + 8 * lg) * 32 + 16 + lh] = (_Float16)p1;
    }

    // Cross-lane LDS RAW inside the wave: force DS completion explicitly.
    asm volatile("s_wait_dscnt 0" ::: "memory");

    // ---- Reload P as A-operand (16x32 f16) ----
    v16h pa;
#pragma unroll
    for (int j = 0; j < 8; ++j) {
      pa[j]     = pb[lh * 32 + lg * 16 + j];
      pa[j + 8] = pb[lh * 32 + lg * 16 + 8 + j];
    }

    // ---- V B-operands: reduction = key index, cols = 16-wide D tiles ----
    const float* vr = vbase + (size_t)(kb + lg * 16) * KVSTRIDE_ + lh;
    v16h bv0, bv1, bv2, bv3;
#pragma unroll
    for (int j = 0; j < 16; ++j) {
      const float* p = vr + (size_t)j * KVSTRIDE_;
      bv0[j] = (_Float16)p[0];
      bv1[j] = (_Float16)p[16];
      bv2[j] = (_Float16)p[32];
      bv3[j] = (_Float16)p[48];
    }

    o0 = __builtin_amdgcn_wmma_f32_16x16x32_f16(false, pa, false, bv0, (short)0, o0, false, false);
    o1 = __builtin_amdgcn_wmma_f32_16x16x32_f16(false, pa, false, bv1, (short)0, o1, false, false);
    o2 = __builtin_amdgcn_wmma_f32_16x16x32_f16(false, pa, false, bv2, (short)0, o2, false, false);
    o3 = __builtin_amdgcn_wmma_f32_16x16x32_f16(false, pa, false, bv3, (short)0, o3, false, false);
  }

  // ---- Normalize and write output (fp32) ----
#pragma unroll
  for (int vv = 0; vv < 8; ++vv) {
    const int row = qbase + vv + 8 * lg;
    const float inv = 1.0f / lrun[vv];
    float* orow = out + (size_t)(b * S_ + row) * (H_ * D_) + h * D_ + lh;
    orow[0]  = o0[vv] * inv;
    orow[16] = o1[vv] * inv;
    orow[32] = o2[vv] * inv;
    orow[48] = o3[vv] * inv;
  }
}

// ---------------------------------------------------------------------------
// Launcher. d_in order: q, k, v, k_cache, v_cache, slot_mapping.
// d_out: [o | k_cache | v_cache] flat fp32.
// ---------------------------------------------------------------------------
extern "C" void kernel_launch(void* const* d_in, const int* in_sizes, int n_in,
                              void* d_out, int out_size, void* d_ws, size_t ws_size,
                              hipStream_t stream) {
  const float* q     = (const float*)d_in[0];
  const float* k     = (const float*)d_in[1];
  const float* v     = (const float*)d_in[2];
  const float* kc_in = (const float*)d_in[3];
  const float* vc_in = (const float*)d_in[4];
  const int*   slot  = (const int*)d_in[5];

  float* o_out  = (float*)d_out;
  float* kc_out = o_out  + (size_t)B_ * S_ * H_ * D_;
  float* vc_out = kc_out + (size_t)NSLOTS_ * KVSTRIDE_;

  const size_t cache_bytes = sizeof(float) * (size_t)NSLOTS_ * KVSTRIDE_;
  hipMemcpyAsync(kc_out, kc_in, cache_bytes, hipMemcpyDeviceToDevice, stream);
  hipMemcpyAsync(vc_out, vc_in, cache_bytes, hipMemcpyDeviceToDevice, stream);

  scatter_kv_kernel<<<B_ * S_, KVSTRIDE_, 0, stream>>>(k, v, slot, kc_out, vc_out);

  attn_fwd_kernel<<<B_ * H_ * (S_ / 128), 256, 0, stream>>>(q, k, v, o_out);
}
